// GATLayer_14594298871998
// MI455X (gfx1250) — matrix operand order repaired and checked by reading, not compiled, not run on hardware
//
#include <hip/hip_runtime.h>

// ---------------------------------------------------------------------------
// GAT layer, fused flash-softmax formulation for MI455X (gfx1250, wave32).
// N=8192, IN_F=128, OUT_F=64.  HBM-bound: one pass over 512 MB of adjacency
// (~22 us at 23.3 TB/s).  attention @ h AND the softmax row-sums are done
// with v_wmma_f32_16x16x32_f16 (f32 accumulate); accumulator rescale is
// skipped via a wave-uniform branch whenever no row max changed.
// ---------------------------------------------------------------------------

#define NN    8192
#define INF_  128
#define OUTF  64
#define ALPHA 0.2f
#define NEG_BIG (-9.0e15f)

typedef __attribute__((ext_vector_type(16))) _Float16 v16h;
typedef __attribute__((ext_vector_type(8)))  float    v8f;
typedef __attribute__((ext_vector_type(4)))  float    f32x4;

// ---------------------------------------------------------------------------
// Kernel 1: h = features @ W^T  (f32), plus f16 transposed copy hT[OUTF][NN]
// laid out so flash-kernel B tiles are contiguous 32B loads.
// ---------------------------------------------------------------------------
__global__ __launch_bounds__(256) void gat_h_kernel(const float* __restrict__ feat,
                                                    const float* __restrict__ W,
                                                    float* __restrict__ h,
                                                    _Float16* __restrict__ hT) {
  int idx = blockIdx.x * 256 + threadIdx.x;   // 8192*64 threads
  int i = idx >> 6;
  int o = idx & 63;
  const f32x4* fr = (const f32x4*)(feat + (size_t)i * INF_);
  const f32x4* wr = (const f32x4*)(W    + (size_t)o * INF_);
  float s = 0.0f;
#pragma unroll
  for (int k = 0; k < INF_ / 4; ++k) {
    f32x4 a = fr[k], b = wr[k];
    s += a.x * b.x + a.y * b.y + a.z * b.z + a.w * b.w;
  }
  h[(size_t)i * OUTF + o]  = s;
  hT[(size_t)o * NN + i]   = (_Float16)s;
}

// ---------------------------------------------------------------------------
// Kernel 2: s_i = h @ a[:64], s_j = h @ a[64:]
// ---------------------------------------------------------------------------
__global__ __launch_bounds__(256) void gat_s_kernel(const float* __restrict__ h,
                                                    const float* __restrict__ a,
                                                    float* __restrict__ si,
                                                    float* __restrict__ sj) {
  int i = blockIdx.x * 256 + threadIdx.x;
  const float* hr = h + (size_t)i * OUTF;
  float x = 0.0f, y = 0.0f;
#pragma unroll
  for (int o = 0; o < OUTF; ++o) {
    float hv = hr[o];
    x += hv * a[o];
    y += hv * a[OUTF + o];
  }
  si[i] = x;
  sj[i] = y;
}

// ---------------------------------------------------------------------------
// Kernel 3: fused mask/leaky/softmax/AV with online softmax.
// Block = 16 rows, 8 waves; each wave scans a 1024-column chunk in 32-col
// tiles.  A-fragment (P tile, 16x32 f16) is built directly in the WMMA
// registers, k = 16*(lane/16) + element (same contiguous k order as B):
//   lane<16:  M=lane,     elements 0..7 -> K=0..7,   8..15 -> K=8..15
//   lane>=16: M=lane-16,  elements 0..7 -> K=16..23,  8..15 -> K=24..31
// B tile (H, 32x16 f16) per lane = 32 contiguous bytes of hT:
//   lane<16: K=0..15 of column n0+lane;  lane>=16: K=16..31 of n0+lane-16.
// C/D 16x16 f32: lane<16 vgpr v -> (M=v, N=lane); lane>=16 -> (M=v+8).
// A 5th WMMA against an all-ones B tile accumulates the softmax row sums
// (l) in the same C layout, so no cross-lane sum reduction is needed.
// ---------------------------------------------------------------------------
__global__ __launch_bounds__(256) void gat_flash_kernel(
    const float* __restrict__ geo, const float* __restrict__ sem,
    const float* __restrict__ si,  const float* __restrict__ sj,
    const _Float16* __restrict__ hT, float* __restrict__ out) {
  constexpr int WAVES = 8;
  constexpr int CHUNK = NN / WAVES;   // 1024 columns per wave

  __shared__ float lds_m[WAVES][16];
  __shared__ float lds_l[WAVES][16];
  __shared__ float lds_acc[WAVES][16][OUTF];   // 32 KB

  const int lane = threadIdx.x & 31;
  const int wave = threadIdx.x >> 5;
  const int r0   = blockIdx.x * 16;
  const int rloc = lane & 15;               // row within block / N within tile
  const int hi   = (lane >> 4) & 1;         // lane half selector
  const int row  = r0 + rloc;

  float m = -3.0e38f;                       // running row max (A layout: lane=row)
  v8f acc0 = {}, acc1 = {}, acc2 = {}, acc3 = {};
  v8f acc4 = {};                            // row sums l (every column identical)

  v16h bones;                               // all-ones B tile for the l-WMMA
#pragma unroll
  for (int k = 0; k < 16; ++k) bones[k] = (_Float16)1.0f;

  const float s_i_row = si[row];
  const f32x4* grow = (const f32x4*)(geo + (size_t)row * NN);
  const f32x4* srow = (const f32x4*)(sem + (size_t)row * NN);

  const int jstart = wave * CHUNK;
  for (int j0 = jstart; j0 < jstart + CHUNK; j0 += 32) {
    const int c0 = j0 + hi * 16;    // first 8-col run for this lane
    const int c1 = c0 + 8;          // second 8-col run

    // Adjacency: the 512MB HBM stream -> non-temporal, keep hT/sj in L2.
    f32x4 g0 = __builtin_nontemporal_load(grow + (c0 >> 2));
    f32x4 g1 = __builtin_nontemporal_load(grow + (c0 >> 2) + 1);
    f32x4 g2 = __builtin_nontemporal_load(grow + (c1 >> 2));
    f32x4 g3 = __builtin_nontemporal_load(grow + (c1 >> 2) + 1);
    f32x4 q0 = __builtin_nontemporal_load(srow + (c0 >> 2));
    f32x4 q1 = __builtin_nontemporal_load(srow + (c0 >> 2) + 1);
    f32x4 q2 = __builtin_nontemporal_load(srow + (c1 >> 2));
    f32x4 q3 = __builtin_nontemporal_load(srow + (c1 >> 2) + 1);
    // B tiles: 32B contiguous per lane from hT (L2 hit) -- issue early
    const _Float16* b0p = hT + (size_t)(0 * 16 + rloc) * NN + j0 + hi * 16;
    const _Float16* b1p = hT + (size_t)(1 * 16 + rloc) * NN + j0 + hi * 16;
    const _Float16* b2p = hT + (size_t)(2 * 16 + rloc) * NN + j0 + hi * 16;
    const _Float16* b3p = hT + (size_t)(3 * 16 + rloc) * NN + j0 + hi * 16;
    v16h b0 = *(const v16h*)b0p;
    v16h b1 = *(const v16h*)b1p;
    v16h b2 = *(const v16h*)b2p;
    v16h b3 = *(const v16h*)b3p;
    // prefetch ~8 tiles ahead in this row (speculative, global_prefetch_b8)
    __builtin_prefetch((const void*)(grow + (c0 >> 2) + 64), 0, 1);
    __builtin_prefetch((const void*)(srow + (c0 >> 2) + 64), 0, 1);

    // s_j (32 KB, L2 resident)
    f32x4 j0v = *(const f32x4*)(sj + c0);
    f32x4 j1v = *(const f32x4*)(sj + c0 + 4);
    f32x4 j2v = *(const f32x4*)(sj + c1);
    f32x4 j3v = *(const f32x4*)(sj + c1 + 4);

    // masked leaky-relu logits, in A-fragment element order
    float att[16];
    auto quad = [&](f32x4 g, f32x4 q, f32x4 s, int base) {
#pragma unroll
      for (int e = 0; e < 4; ++e) {
        float c  = g[e] + q[e];                       // combined_adj
        float x  = s_i_row + s[e];
        float lr = fmaxf(x, ALPHA * x);               // leaky_relu (ALPHA<1)
        att[base + e] = (c > 0.0f) ? lr * c : NEG_BIG;
      }
    };
    quad(g0, q0, j0v, 0);
    quad(g1, q1, j1v, 4);
    quad(g2, q2, j2v, 8);
    quad(g3, q3, j3v, 12);

    // online softmax: tile max -> pair-reduce across the two lane halves
    float tmax = att[0];
#pragma unroll
    for (int k = 1; k < 16; ++k) tmax = fmaxf(tmax, att[k]);
    tmax = fmaxf(tmax, __shfl_xor(tmax, 16, 32));
    float m_new = fmaxf(m, tmax);

    // rescale accumulators only if any row max actually increased.
    // __any() is wave-uniform -> scalar branch, EXEC stays all-ones.
    if (__any(m_new > m)) {
      float scale = __expf(m - m_new);
#pragma unroll
      for (int v = 0; v < 8; ++v) {
        float sv = __shfl(scale, v + hi * 8, 32);  // row v(+8)'s scale
        acc0[v] *= sv; acc1[v] *= sv; acc2[v] *= sv; acc3[v] *= sv;
        acc4[v] *= sv;
      }
    }
    m = m_new;

    v16h afrag;
#pragma unroll
    for (int k = 0; k < 16; ++k) {
      afrag[k] = (_Float16)__expf(att[k] - m);
    }

    acc0 = __builtin_amdgcn_wmma_f32_16x16x32_f16(false, afrag, false, b0,
                                                  (short)0, acc0, false, false);
    acc1 = __builtin_amdgcn_wmma_f32_16x16x32_f16(false, afrag, false, b1,
                                                  (short)0, acc1, false, false);
    acc2 = __builtin_amdgcn_wmma_f32_16x16x32_f16(false, afrag, false, b2,
                                                  (short)0, acc2, false, false);
    acc3 = __builtin_amdgcn_wmma_f32_16x16x32_f16(false, afrag, false, b3,
                                                  (short)0, acc3, false, false);
    acc4 = __builtin_amdgcn_wmma_f32_16x16x32_f16(false, afrag, false, bones,
                                                  (short)0, acc4, false, false);
  }

  // park wave partials in LDS
  if (lane < 16) {
    lds_m[wave][lane] = m;
  }
  if (rloc == 0) {   // acc4 columns are identical; one column writes l
#pragma unroll
    for (int v = 0; v < 8; ++v) lds_l[wave][v + hi * 8] = acc4[v];
  }
#pragma unroll
  for (int v = 0; v < 8; ++v) {
    int rr = v + hi * 8;
    lds_acc[wave][rr][0 * 16 + rloc] = acc0[v];
    lds_acc[wave][rr][1 * 16 + rloc] = acc1[v];
    lds_acc[wave][rr][2 * 16 + rloc] = acc2[v];
    lds_acc[wave][rr][3 * 16 + rloc] = acc3[v];
  }
  __syncthreads();

  // merge the 8 wave partials (flash combine) + normalize + ELU + store
  const int tid = threadIdx.x;
  const int c = tid & 63;
#pragma unroll
  for (int rr4 = 0; rr4 < 4; ++rr4) {
    int r = (tid >> 6) + rr4 * 4;
    float mstar = lds_m[0][r];
#pragma unroll
    for (int w = 1; w < WAVES; ++w) mstar = fmaxf(mstar, lds_m[w][r]);
    float num = 0.0f, den = 0.0f;
#pragma unroll
    for (int w = 0; w < WAVES; ++w) {
      float f = __expf(lds_m[w][r] - mstar);
      num += lds_acc[w][r][c] * f;
      den += lds_l[w][r] * f;
    }
    float v = num / den;
    out[(size_t)(r0 + r) * OUTF + c] = (v > 0.0f) ? v : (__expf(v) - 1.0f);
  }
}

// ---------------------------------------------------------------------------
extern "C" void kernel_launch(void* const* d_in, const int* in_sizes, int n_in,
                              void* d_out, int out_size, void* d_ws, size_t ws_size,
                              hipStream_t stream) {
  const float* geo  = (const float*)d_in[0];
  const float* sem  = (const float*)d_in[1];
  const float* feat = (const float*)d_in[2];
  const float* W    = (const float*)d_in[3];
  const float* a    = (const float*)d_in[4];
  float* out = (float*)d_out;

  char* ws = (char*)d_ws;
  float*    h  = (float*)ws;                                    // 2 MB
  _Float16* hT = (_Float16*)(ws + (size_t)NN * OUTF * 4);       // 1 MB
  float*    si = (float*)(ws + (size_t)NN * OUTF * 6);          // 32 KB
  float*    sj = si + NN;                                       // 32 KB

  gat_h_kernel<<<(NN * OUTF) / 256, 256, 0, stream>>>(feat, W, h, hT);
  gat_s_kernel<<<NN / 256, 256, 0, stream>>>(h, a, si, sj);
  gat_flash_kernel<<<NN / 16, 256, 0, stream>>>(geo, sem, si, sj, hT, out);
}
